// RelativeGlobalAttention_32272384262656
// MI455X (gfx1250) — matrix operand order repaired and checked by reading, not compiled, not run on hardware
//
#include <hip/hip_runtime.h>
#include <hip/hip_bf16.h>

#define B_  4
#define L_  2048
#define D_  512
#define DK_ 256

typedef __bf16 bf16_t;
typedef __attribute__((ext_vector_type(16))) __bf16 v16bf;
typedef __attribute__((ext_vector_type(8)))  __bf16 v8bf;
typedef __attribute__((ext_vector_type(8)))  float  v8f;

// ---------- helpers ----------

static __device__ __forceinline__ bf16_t f2bf(float x) {
  // round-to-nearest-even fp32 -> bf16
  unsigned u = __float_as_uint(x);
  unsigned r = (u + 0x7FFFu + ((u >> 16) & 1u)) >> 16;
  unsigned short h = (unsigned short)r;
  return __builtin_bit_cast(bf16_t, h);
}

static __device__ __forceinline__ v8f wmma_bf16(v16bf a, v16bf b, v8f c) {
  // D = A(16x32 bf16) * B(32x16 bf16) + C(16x16 f32)
  return __builtin_amdgcn_wmma_f32_16x16x32_bf16(
      /*neg_a=*/false, a, /*neg_b=*/false, b,
      /*c_mod=*/(short)0, c, /*reuse_a=*/false, /*reuse_b=*/false);
}

// A-matrix 16x32 bf16 fragment from row-major source (base = &A[row0][k0]).
// Lane r<16: row r, K = {0..7, 16..23}; lane r+16: row r, K = {8..15, 24..31}.
// Each half is 8 contiguous bf16 -> two global_load_b128 per lane.
static __device__ __forceinline__ v16bf load_a_bf16(const bf16_t* __restrict__ base, int ld) {
  const int lane = threadIdx.x & 31;
  const int r  = lane & 15;
  const int kh = lane >> 4;
  const bf16_t* p = base + (size_t)r * ld + kh * 8;
  v8bf lo = *(const v8bf*)(p);        // K = kh*8 .. kh*8+7
  v8bf hi = *(const v8bf*)(p + 16);   // K = 16+kh*8 .. 16+kh*8+7
  return __builtin_shufflevector(lo, hi, 0, 1, 2, 3, 4, 5, 6, 7,
                                 8, 9, 10, 11, 12, 13, 14, 15);
}

// B-matrix 32x16 where B[k][n] = S[n][k], S row-major (base = &S[n0][k0]).
// Lane n (half kg) reads K = {kg*8..kg*8+7, 16+kg*8..16+kg*8+7} of row n -> two b128 loads.
static __device__ __forceinline__ v16bf load_bt_bf16(const bf16_t* __restrict__ base, int ld) {
  const int lane = threadIdx.x & 31;
  const int n  = lane & 15;
  const int kg = lane >> 4;
  const bf16_t* p = base + (size_t)n * ld + kg * 8;
  v8bf lo = *(const v8bf*)(p);
  v8bf hi = *(const v8bf*)(p + 16);
  return __builtin_shufflevector(lo, hi, 0, 1, 2, 3, 4, 5, 6, 7,
                                 8, 9, 10, 11, 12, 13, 14, 15);
}

// ---------- kernels ----------

__global__ void cvt_f32_bf16(const float* __restrict__ src, bf16_t* __restrict__ dst, size_t n) {
  size_t i = (size_t)blockIdx.x * blockDim.x + threadIdx.x;
  size_t stride = (size_t)gridDim.x * blockDim.x;
  for (; i < n; i += stride) dst[i] = f2bf(src[i]);
}

// W[K,N] f32 -> WT[N,K] bf16, LDS-tiled so both sides are coalesced.
// grid(K/32, N/32), block 256 (32x8).
__global__ __launch_bounds__(256) void cvt_transpose_f32_bf16(
    const float* __restrict__ src, bf16_t* __restrict__ dst, int K, int N) {
  __shared__ float tile[32][33];
  const int k0 = blockIdx.x * 32, n0 = blockIdx.y * 32;
  const int tx = threadIdx.x & 31, ty = threadIdx.x >> 5;
#pragma unroll
  for (int i = ty; i < 32; i += 8)
    tile[i][tx] = src[(size_t)(k0 + i) * N + n0 + tx];
  __syncthreads();
#pragma unroll
  for (int i = ty; i < 32; i += 8)
    dst[(size_t)(n0 + i) * K + k0 + tx] = f2bf(tile[tx][i]);
}

// V[B,L,D] bf16 -> VT[B,D,L] bf16, LDS-tiled. grid(L/32, D/32, B), block 256.
__global__ __launch_bounds__(256) void transpose_bf16(
    const bf16_t* __restrict__ src, bf16_t* __restrict__ dst) {
  __shared__ bf16_t tile[32][33];
  const int b = blockIdx.z;
  const int l0 = blockIdx.x * 32, d0 = blockIdx.y * 32;
  const int tx = threadIdx.x & 31, ty = threadIdx.x >> 5;
#pragma unroll
  for (int i = ty; i < 32; i += 8)
    tile[i][tx] = src[((size_t)b * L_ + l0 + i) * D_ + d0 + tx];
  __syncthreads();
#pragma unroll
  for (int i = ty; i < 32; i += 8)
    dst[((size_t)b * D_ + d0 + i) * L_ + l0 + tx] = tile[tx][i];
}

// WMMA GEMM with transposed B operand: C[M,N] = A[M,K] * Bt^T,
// where Bt is row-major [N,K] (so B[k][n] = Bt[n][k]; all loads contiguous).
// OUTBF: bf16 vs f32 output. block = 128 (4 waves); 32x32 C tile per wave.
template <bool OUTBF>
__global__ __launch_bounds__(128) void gemm_wmma_bt(
    const bf16_t* __restrict__ A, const bf16_t* __restrict__ Bt,
    float* __restrict__ Cf, bf16_t* __restrict__ Cb,
    int M, int N, int K, int lda, int ldb, int ldc) {
  const int lane = threadIdx.x & 31;
  const int wave = threadIdx.x >> 5;
  const int m0 = (blockIdx.x * 4 + wave) * 32;
  const int n0 = blockIdx.y * 32;
  if (m0 >= M || n0 >= N) return;

  v8f acc00 = {}, acc01 = {}, acc10 = {}, acc11 = {};
  for (int k0 = 0; k0 < K; k0 += 32) {
    v16bf a0 = load_a_bf16(A + (size_t)m0 * lda + k0, lda);
    v16bf a1 = load_a_bf16(A + (size_t)(m0 + 16) * lda + k0, lda);
    v16bf b0 = load_bt_bf16(Bt + (size_t)n0 * ldb + k0, ldb);
    v16bf b1 = load_bt_bf16(Bt + (size_t)(n0 + 16) * ldb + k0, ldb);
    acc00 = wmma_bf16(a0, b0, acc00);
    acc01 = wmma_bf16(a0, b1, acc01);
    acc10 = wmma_bf16(a1, b0, acc10);
    acc11 = wmma_bf16(a1, b1, acc11);
  }

  const int rb = (lane >> 4) * 8;
  const int cc = lane & 15;
#pragma unroll
  for (int v = 0; v < 8; ++v) {
    size_t r0 = (size_t)(m0 + rb + v) * ldc;
    size_t r1 = (size_t)(m0 + 16 + rb + v) * ldc;
    if constexpr (OUTBF) {
      Cb[r0 + n0 + cc]      = f2bf(acc00[v]);
      Cb[r0 + n0 + 16 + cc] = f2bf(acc01[v]);
      Cb[r1 + n0 + cc]      = f2bf(acc10[v]);
      Cb[r1 + n0 + 16 + cc] = f2bf(acc11[v]);
    } else {
      Cf[r0 + n0 + cc]      = acc00[v];
      Cf[r0 + n0 + 16 + cc] = acc01[v];
      Cf[r1 + n0 + cc]      = acc10[v];
      Cf[r1 + n0 + 16 + cc] = acc11[v];
    }
  }
}

// Skew gather: Srel[l,m] from precomputed QE[B,L,L].
static __device__ __forceinline__ float srel_val(const float* __restrict__ QE,
                                                 int b, int l, int m) {
  size_t base = (size_t)b * L_ * L_;
  if (m <= l)     return QE[base + (size_t)l * L_ + (L_ - 1 - l + m)];
  if (m == l + 1) return 0.f;
  return QE[base + (size_t)(l + 1) * L_ + (m - l - 2)];
}

// Fused flash attention with relative-position skew.
// grid = (L/16, B); block = 128 (4 waves). Each workgroup: 16 query rows.
// Wave w owns output columns [w*128, w*128+128). V is pre-transposed [B,D,L].
__global__ __launch_bounds__(128) void attn_kernel(
    const bf16_t* __restrict__ Qb, const bf16_t* __restrict__ Kb,
    const bf16_t* __restrict__ VT, const float* __restrict__ QE,
    float* __restrict__ out) {
  __shared__ __align__(128) float pbuf[4][16 * 32];  // per-wave P staging

  const int lane  = threadIdx.x & 31;
  const int wave  = threadIdx.x >> 5;
  const int b     = blockIdx.y;
  const int l0    = blockIdx.x * 16;
  const int c0    = wave * 128;
  const int col   = lane & 15;
  const int rhalf = lane >> 4;

  // Q tile (16 rows x 256) kept in registers as 8 A-fragments.
  v16bf qf[8];
  const bf16_t* qbase = Qb + ((size_t)b * L_ + l0) * DK_;
#pragma unroll
  for (int kk = 0; kk < 8; ++kk) qf[kk] = load_a_bf16(qbase + kk * 32, DK_);

  v8f acc[8];
#pragma unroll
  for (int c = 0; c < 8; ++c) acc[c] = (v8f){};

  float mrow[8], lrow[8];
#pragma unroll
  for (int v = 0; v < 8; ++v) { mrow[v] = -1e30f; lrow[v] = 0.f; }

  const float sc = 0.0625f;  // 1/sqrt(256)
  float* pb = pbuf[wave];

  for (int m0 = 0; m0 < L_; m0 += 32) {
    // ---- scores: S[16x32] = Q(16x256) * K^T (contiguous B^T loads) ----
    v8f s0 = {}, s1 = {};
    const bf16_t* kbase = Kb + ((size_t)b * L_ + m0) * DK_;
    if (m0 + 32 < L_)
      __builtin_prefetch((const void*)(kbase + 32 * DK_), 0, 1);
#pragma unroll
    for (int kk = 0; kk < 8; ++kk) {
      v16bf kf0 = load_bt_bf16(kbase + kk * 32, DK_);
      v16bf kf1 = load_bt_bf16(kbase + (size_t)16 * DK_ + kk * 32, DK_);
      s0 = wmma_bf16(qf[kk], kf0, s0);
      s1 = wmma_bf16(qf[kk], kf1, s1);
    }

    // ---- add skewed relative logits, scale ----
#pragma unroll
    for (int v = 0; v < 8; ++v) {
      int l = l0 + rhalf * 8 + v;
      s0[v] = (s0[v] + srel_val(QE, b, l, m0 + col)) * sc;
      s1[v] = (s1[v] + srel_val(QE, b, l, m0 + 16 + col)) * sc;
    }

    // ---- online softmax (row reductions across 16-lane half-wave) ----
    float alpha[8];
#pragma unroll
    for (int v = 0; v < 8; ++v) {
      float t = fmaxf(s0[v], s1[v]);
#pragma unroll
      for (int off = 1; off < 16; off <<= 1) t = fmaxf(t, __shfl_xor(t, off, 32));
      float mn = fmaxf(mrow[v], t);
      float a  = __expf(mrow[v] - mn);
      float p0 = __expf(s0[v] - mn);
      float p1 = __expf(s1[v] - mn);
      s0[v] = p0; s1[v] = p1;
      float rs = p0 + p1;
#pragma unroll
      for (int off = 1; off < 16; off <<= 1) rs += __shfl_xor(rs, off, 32);
      lrow[v] = lrow[v] * a + rs;
      mrow[v] = mn;
      alpha[v] = a;
    }
#pragma unroll
    for (int c = 0; c < 8; ++c)
#pragma unroll
      for (int v = 0; v < 8; ++v) acc[c][v] *= alpha[v];

    // ---- P: C-layout f32 -> A-layout bf16 via LDS bounce ----
#pragma unroll
    for (int v = 0; v < 8; ++v) {
      int lr = rhalf * 8 + v;
      pb[lr * 32 + col]      = s0[v];
      pb[lr * 32 + 16 + col] = s1[v];
    }
    __syncthreads();
    v16bf pA;
    {
      int r  = lane & 15;
      int kh = lane >> 4;
      const float* q0 = pb + r * 32 + kh * 8;        // K = kh*8 .. +7
      v8f x0 = *(const v8f*)(q0);                    // ds_load_b128 x2
      v8f x1 = *(const v8f*)(q0 + 16);               // K = 16+kh*8 .. +7
#pragma unroll
      for (int e = 0; e < 8; ++e) {
        pA[e]     = f2bf(x0[e]);
        pA[8 + e] = f2bf(x1[e]);
      }
    }
    __syncthreads();

    // ---- accumulate O += P(16x32) * V(32x128-slice), V^T contiguous ----
    const bf16_t* vtb = VT + (size_t)b * D_ * L_ + (size_t)c0 * L_ + m0;
#pragma unroll
    for (int c = 0; c < 8; ++c) {
      v16bf vf = load_bt_bf16(vtb + (size_t)c * 16 * L_, L_);
      acc[c] = wmma_bf16(pA, vf, acc[c]);
    }
  }

  // ---- epilogue: divide by row sums, store fp32 ----
#pragma unroll
  for (int v = 0; v < 8; ++v) {
    float inv = 1.f / lrow[v];
    int l = l0 + rhalf * 8 + v;
    float* orow = out + ((size_t)b * L_ + l) * D_ + c0;
#pragma unroll
    for (int c = 0; c < 8; ++c) orow[c * 16 + col] = acc[c][v] * inv;
  }
}

// ---------- launch ----------

extern "C" void kernel_launch(void* const* d_in, const int* in_sizes, int n_in,
                              void* d_out, int out_size, void* d_ws, size_t ws_size,
                              hipStream_t stream) {
  const float* inQ = (const float*)d_in[0];
  const float* inK = (const float*)d_in[1];
  const float* inV = (const float*)d_in[2];
  const float* Wq  = (const float*)d_in[3];
  const float* Wk  = (const float*)d_in[4];
  const float* Wv  = (const float*)d_in[5];
  const float* E   = (const float*)d_in[6];
  float* out = (float*)d_out;

  char* ws = (char*)d_ws;
  size_t off = 0;
  auto alloc = [&](size_t bytes) -> char* {
    char* p = ws + off;
    off += (bytes + 255) & ~(size_t)255;
    return p;
  };

  const size_t nIn = (size_t)B_ * L_ * D_;      // 4,194,304
  const size_t nPK = (size_t)B_ * L_ * DK_;     // 2,097,152

  bf16_t* bQin = (bf16_t*)alloc(nIn * 2);
  bf16_t* bKin = (bf16_t*)alloc(nIn * 2);
  bf16_t* bVin = (bf16_t*)alloc(nIn * 2);
  bf16_t* bWqT = (bf16_t*)alloc((size_t)D_ * DK_ * 2);   // [DK, D]
  bf16_t* bWkT = (bf16_t*)alloc((size_t)D_ * DK_ * 2);   // [DK, D]
  bf16_t* bWvT = (bf16_t*)alloc((size_t)D_ * D_ * 2);    // [D, D]
  bf16_t* bE   = (bf16_t*)alloc((size_t)L_ * DK_ * 2);   // [L, DK] (already B^T form)
  bf16_t* Qb   = (bf16_t*)alloc(nPK * 2);                // [B*L, DK]
  bf16_t* Kb   = (bf16_t*)alloc(nPK * 2);                // [B*L, DK]
  bf16_t* Vb   = (bf16_t*)alloc(nIn * 2);                // [B, L, D]
  bf16_t* VbT  = (bf16_t*)alloc(nIn * 2);                // [B, D, L]
  float*  QEf  = (float*)alloc((size_t)B_ * L_ * L_ * 4);  // 64 MB

  // fp32 -> bf16 conversions (weights transposed for contiguous B^T loads)
  cvt_f32_bf16<<<2048, 256, 0, stream>>>(inQ, bQin, nIn);
  cvt_f32_bf16<<<2048, 256, 0, stream>>>(inK, bKin, nIn);
  cvt_f32_bf16<<<2048, 256, 0, stream>>>(inV, bVin, nIn);
  cvt_transpose_f32_bf16<<<dim3(D_ / 32, DK_ / 32), 256, 0, stream>>>(Wq, bWqT, D_, DK_);
  cvt_transpose_f32_bf16<<<dim3(D_ / 32, DK_ / 32), 256, 0, stream>>>(Wk, bWkT, D_, DK_);
  cvt_transpose_f32_bf16<<<dim3(D_ / 32, D_ / 32), 256, 0, stream>>>(Wv, bWvT, D_, D_);
  cvt_f32_bf16<<<512, 256, 0, stream>>>(E, bE, (size_t)L_ * DK_);

  const int M = B_ * L_;  // 8192 rows, batch folded into M (weights shared)

  // Q = inQ*Wq, K = inK*Wk, V = inV*Wv (bf16 out, row-major)
  gemm_wmma_bt<true><<<dim3(M / 128, DK_ / 32), 128, 0, stream>>>(
      bQin, bWqT, nullptr, Qb, M, DK_, D_, D_, D_, DK_);
  gemm_wmma_bt<true><<<dim3(M / 128, DK_ / 32), 128, 0, stream>>>(
      bKin, bWkT, nullptr, Kb, M, DK_, D_, D_, D_, DK_);
  gemm_wmma_bt<true><<<dim3(M / 128, D_ / 32), 128, 0, stream>>>(
      bVin, bWvT, nullptr, Vb, M, D_, D_, D_, D_, D_);

  // V -> V^T per batch for contiguous P*V fragment loads
  transpose_bf16<<<dim3(L_ / 32, D_ / 32, B_), 256, 0, stream>>>(Vb, VbT);

  // QE = Q * E^T  (fp32 out, [B*L, L]); E is row-major [L, DK] = B^T form
  gemm_wmma_bt<false><<<dim3(M / 128, L_ / 32), 128, 0, stream>>>(
      Qb, bE, QEf, nullptr, M, L_, DK_, DK_, DK_, L_);

  // fused QK^T + skew + softmax + attn*V
  attn_kernel<<<dim3(L_ / 16, B_), 128, 0, stream>>>(Qb, Kb, VbT, QEf, out);
}
